// RNNDecoder_8022998909784
// MI455X (gfx1250) — matrix operand order repaired and checked by reading, not compiled, not run on hardware
//
#include <hip/hip_runtime.h>
#include <hip/hip_bf16.h>
#include <stdint.h>

// ---------------------------------------------------------------------------
// GRU decoder for MI455X (gfx1250, wave32).
//  * all matmuls via v_wmma_f32_16x16x32_bf16 (fp32 accumulate)
//  * weights pre-cast to bf16 in workspace (one pass)
//  * GEMM: A staged in LDS as 64x128 double-buffered panels via
//    GLOBAL_LOAD_ASYNC_TO_LDS_B128 (ASYNCcnt); inner 4 K-substeps are
//    barrier-free so the B stream pipelines as global_load_b128 clauses
//  * recurrence: one persistent kernel per layer; the 192KB W_hh slice is
//    DMA'd into LDS by the Tensor Data Mover (TENSOR_LOAD_TO_LDS, 3 x 2D
//    descriptors with LDS padding), then 128 timesteps run out of LDS with a
//    global-atomic step barrier
//  * output projection streams 524MB of logits with nontemporal stores
// ---------------------------------------------------------------------------

typedef unsigned short u16x;                                   // bf16 bits
typedef __attribute__((ext_vector_type(16))) __bf16 v16bf;
typedef __attribute__((ext_vector_type(8)))  float  v8f;
typedef __attribute__((ext_vector_type(4)))  unsigned u32x4v;  // TDM D# group0
typedef __attribute__((ext_vector_type(8)))  unsigned u32x8v;  // TDM D# group1

#define WMMA_BF16(a, b, c) \
  __builtin_amdgcn_wmma_f32_16x16x32_bf16(false, (a), false, (b), (short)0, (c), false, false)

__device__ __forceinline__ u16x f32_to_bf16(float f) {
  union { float f; unsigned u; } v; v.f = f;
  unsigned r = v.u + 0x7FFFu + ((v.u >> 16) & 1u);             // round-nearest-even
  return (u16x)(r >> 16);
}

// Async global->LDS copy of one 16B chunk per lane (ASYNCcnt-tracked).
__device__ __forceinline__ void async_b128_to_lds(const u16x* gaddr, const u16x* laddr) {
  unsigned ldsOff = (unsigned)(uintptr_t)laddr;     // low 32 bits = LDS byte address
  asm volatile("global_load_async_to_lds_b128 %0, %1, off"
               :: "v"(ldsOff), "v"(gaddr) : "memory");
}
__device__ __forceinline__ void wait_async4() {
  asm volatile("s_wait_asynccnt 0x4" ::: "memory");
}
__device__ __forceinline__ void wait_tensor0() {
#if __has_builtin(__builtin_amdgcn_s_wait_tensorcnt)
  __builtin_amdgcn_s_wait_tensorcnt(0);
#else
  asm volatile("s_wait_tensorcnt 0x0" ::: "memory");
#endif
}

// A fragment (16 rows x 32 K, bf16): lanes 0-15 hold K = k0..k0+15,
// lanes 16-31 hold K = k0+16..k0+31 (matches load_frag_b).
__device__ __forceinline__ v16bf load_frag_a(const u16x* __restrict__ base, int ld,
                                             int m0, int k0, int lane) {
  const int lh = lane >> 4, m = lane & 15;
  const u16x* p = base + (size_t)(m0 + m) * ld + k0 + lh * 16;
  union { uint4 q[2]; v16bf v; } u;
  u.q[0] = *(const uint4*)(p);
  u.q[1] = *(const uint4*)(p + 8);
  return u.v;
}

// B fragment (32 K x 16 cols) from row-major W[N,K]: lane n supplies column n,
// lanes 0-15: K = k0..k0+15, lanes 16-31: K = k0+16..k0+31.
__device__ __forceinline__ v16bf load_frag_b(const u16x* __restrict__ base, int ld,
                                             int n0, int k0, int lane) {
  const int lh = lane >> 4, n = lane & 15;
  const u16x* p = base + (size_t)(n0 + n) * ld + k0 + lh * 16;
  union { uint4 q[2]; v16bf v; } u;
  u.q[0] = *(const uint4*)(p);
  u.q[1] = *(const uint4*)(p + 8);
  return u.v;
}

// ---------------------------------------------------------------------------
// C[M,N] (fp32) = A[M,K](bf16) * W[N,K](bf16)^T + bias.
// Block 256 thr = 8 waves, BM=64 BN=128, each wave a 32x32 tile (4 x v8f).
// A staged as 64x128 double-buffered LDS panels via async loads (4 b128/thr
// per stage, s_wait_asynccnt 4); inner 4 K-substeps barrier-free. B direct.
// remapB>0: row (t*remapB+b) scattered to (b*remapS+t) with nontemporal stores.
// ---------------------------------------------------------------------------
__global__ void __launch_bounds__(256)
gemm_bf16(const u16x* __restrict__ A, const u16x* __restrict__ Bw,
          const float* __restrict__ bias, float* __restrict__ C,
          int M, int N, int K, int remapB, int remapS) {
  extern __shared__ u16x atile[];          // [2][64][136] bf16 (rows padded)
  const int KC = 128;
  const int LDA = KC + 8;                  // 68 dwords % 64 banks -> conflict-free
  const int tid = threadIdx.x;
  const int lane = tid & 31;
  const int wave = tid >> 5;
  const int wr = wave >> 2;                // 0..1
  const int wc = wave & 3;                 // 0..3
  const int mb = blockIdx.y * 64;
  const int m0 = mb + wr * 32;
  const int n0 = blockIdx.x * 128 + wc * 32;

  // staging: thread covers row = tid/4, segments (tid%4)*8 + i*32, i=0..3
  const int srow = tid >> 2, scol0 = (tid & 3) * 8;
  const u16x* aRow = A + (size_t)(mb + srow) * K + scol0;
  u16x* lRow = atile + srow * LDA + scol0;

  auto stage = [&](int c, int buf) {
    const u16x* g = aRow + c * KC;
    u16x* l = lRow + buf * (64 * LDA);
#pragma unroll
    for (int i = 0; i < 4; ++i)
      async_b128_to_lds(g + i * 32, l + i * 32);
  };

  v8f acc00 = {}, acc01 = {}, acc10 = {}, acc11 = {};
  const int nc = K / KC;
  stage(0, 0);
  for (int c = 0; c < nc; ++c) {
    stage((c + 1 < nc) ? c + 1 : c, (c + 1) & 1);      // last iter: dummy re-stage
    wait_async4();                                      // panel c resident
    __syncthreads();

    const u16x* abase = atile + (c & 1) * (64 * LDA);
    const int lh = lane >> 4, lm = lane & 15;
#pragma unroll
    for (int ks = 0; ks < 4; ++ks) {
      const int k0 = c * KC + ks * 32;
      union { uint4 q[2]; v16bf v; } a0u, a1u;
      const u16x* pa0 = abase + (wr * 32 + lm) * LDA + ks * 32 + lh * 16;
      const u16x* pa1 = pa0 + 16 * LDA;
      a0u.q[0] = *(const uint4*)(pa0); a0u.q[1] = *(const uint4*)(pa0 + 8);
      a1u.q[0] = *(const uint4*)(pa1); a1u.q[1] = *(const uint4*)(pa1 + 8);
      v16bf b0 = load_frag_b(Bw, K, n0,      k0, lane);
      v16bf b1 = load_frag_b(Bw, K, n0 + 16, k0, lane);
      acc00 = WMMA_BF16(a0u.v, b0, acc00);
      acc01 = WMMA_BF16(a0u.v, b1, acc01);
      acc10 = WMMA_BF16(a1u.v, b0, acc10);
      acc11 = WMMA_BF16(a1u.v, b1, acc11);
    }
    __syncthreads();                                    // protect buffer reuse
  }

  const int lh = lane >> 4, ln = lane & 15;
#pragma unroll
  for (int i = 0; i < 2; ++i) {
#pragma unroll
    for (int j = 0; j < 2; ++j) {
      const v8f acc = (i == 0) ? (j == 0 ? acc00 : acc01)
                               : (j == 0 ? acc10 : acc11);
      const int col = n0 + j * 16 + ln;
      const float bv = bias ? bias[col] : 0.0f;
#pragma unroll
      for (int r = 0; r < 8; ++r) {
        int row = m0 + i * 16 + (lh ? 8 + r : r);
        float v = acc[r] + bv;
        if (remapB) {
          int t = row / remapB, b = row % remapB;
          __builtin_nontemporal_store(v, &C[(size_t)(b * remapS + t) * N + col]);
        } else {
          C[(size_t)row * N + col] = v;
        }
      }
    }
  }
}

// ---------------------------------------------------------------------------
// Persistent GRU layer: grid = H/32 blocks of 128 thr (4 waves).
// Block owns hidden columns [j0, j0+32). Its 96x1024 bf16 W_hh slice is DMA'd
// into LDS by the TDM: one 2D descriptor per gate (32 rows x 2048B), with
// pad_amount=4dw every 256dw so LDS rows land at stride 1040 elements.
// Steps synchronized with a global atomic counter (ping-pong h buffers).
// ---------------------------------------------------------------------------
__global__ void __launch_bounds__(128)
gru_layer_persistent(const u16x* __restrict__ Whh,    // [3H, H] bf16
                     const float* __restrict__ bhh,   // [3H]
                     const float* __restrict__ gi,    // [T, 32, 3H] fp32
                     float* __restrict__ h0f, float* __restrict__ h1f,
                     u16x* __restrict__ h0b, u16x* __restrict__ h1b,
                     u16x* __restrict__ y,            // [T, 32, H] bf16
                     unsigned* __restrict__ syncc,
                     int T, int H) {
  extern __shared__ u16x lds[];                       // [96][1040]
  const int LD = 1040;                                // 1024 + 2 pads of 8 elems
  const int lane = threadIdx.x & 31;
  const int wave = threadIdx.x >> 5;
  const int rt = wave >> 1;                           // batch tile 0..1
  const int ct = wave & 1;                            // col tile 0..1
  const int j0 = blockIdx.x * 32;

  // ---- TDM: stage W_hh rows {g*H+j0 .. +31 | g=0..2} into LDS (wave 0) ----
  if (threadIdx.x < 32) {
    const unsigned ldsBase = (unsigned)(uintptr_t)(&lds[0]);
#pragma unroll
    for (int g = 0; g < 3; ++g) {
      unsigned long long ga =
          (unsigned long long)(uintptr_t)(Whh + (size_t)(g * H + j0) * H);
      u32x4v d0;
      d0[0] = 1u;                                        // count=1 (valid D#)
      d0[1] = ldsBase + (unsigned)(g * 32 * LD * 2);     // lds_addr (bytes)
      d0[2] = (unsigned)(ga & 0xFFFFFFFFu);              // global_addr[31:0]
      d0[3] = (unsigned)((ga >> 32) & 0x1FFFFFFu)        // global_addr[56:32]
            | (2u << 30);                                // type=2 (image)
      u32x8v d1;
      d1[0] = (1u << 16)                                 // data_size: 2 bytes
            | (1u << 20)                                 // pad_enable
            | (7u << 22)                                 // pad_interval: 256 dw
            | (3u << 25);                                // pad_amount: 4 dw
      d1[1] = (1024u & 0xFFFFu) << 16;                   // tensor_dim0 = 1024
      d1[2] = ((unsigned)(3 * H) & 0xFFFFu) << 16;       // tensor_dim1 = 3H
      d1[3] = (1024u << 16);                             // tile_dim0 = 1024
      d1[4] = 32u;                                       // tile_dim1 = 32
      d1[5] = (unsigned)H;                               // dim0_stride = H
      d1[6] = 0u;
      d1[7] = 0u;
      asm volatile("tensor_load_to_lds %0, %1" :: "s"(d0), "s"(d1) : "memory");
    }
    wait_tensor0();
  }
  __syncthreads();

  const int lh = lane >> 4, ln = lane & 15;
  const int col  = j0 + ct * 16 + ln;                 // hidden index
  const float bh_r = bhh[col], bh_z = bhh[H + col], bh_n = bhh[2 * H + col];

  for (int t = 0; t < T; ++t) {
    const float* hf_in  = (t & 1) ? h1f : h0f;
    float*       hf_out = (t & 1) ? h0f : h1f;
    const u16x*  hb_in  = (t & 1) ? h1b : h0b;
    u16x*        hb_out = (t & 1) ? h0b : h1b;
    const float* gi_t = gi + (size_t)t * 32 * 3 * H;

    v8f accr = {}, accz = {}, accn = {};
#pragma unroll 2
    for (int k0 = 0; k0 < H; k0 += 32) {
      const int kadj = (k0 >= 512) ? 8 : 0;           // skip TDM mid-row pad
      v16bf a = load_frag_a(hb_in, H, rt * 16, k0, lane);
      union { uint4 q[2]; v16bf v; } b0, b1, b2;      // B frags from LDS
      const u16x* p0 = lds + (0 * 32 + ct * 16 + ln) * LD + k0 + kadj + lh * 16;
      const u16x* p1 = lds + (1 * 32 + ct * 16 + ln) * LD + k0 + kadj + lh * 16;
      const u16x* p2 = lds + (2 * 32 + ct * 16 + ln) * LD + k0 + kadj + lh * 16;
      b0.q[0] = *(const uint4*)(p0); b0.q[1] = *(const uint4*)(p0 + 8);
      b1.q[0] = *(const uint4*)(p1); b1.q[1] = *(const uint4*)(p1 + 8);
      b2.q[0] = *(const uint4*)(p2); b2.q[1] = *(const uint4*)(p2 + 8);
      accr = WMMA_BF16(a, b0.v, accr);
      accz = WMMA_BF16(a, b1.v, accz);
      accn = WMMA_BF16(a, b2.v, accn);
    }

#pragma unroll
    for (int r = 0; r < 8; ++r) {
      int row = rt * 16 + (lh ? 8 + r : r);           // batch index 0..31
      const float* girow = gi_t + (size_t)row * 3 * H;
      float gr = accr[r] + bh_r + girow[col];
      float gz = accz[r] + bh_z + girow[H + col];
      float rg = 1.0f / (1.0f + __expf(-gr));
      float zg = 1.0f / (1.0f + __expf(-gz));
      float ng = tanhf(girow[2 * H + col] + rg * (accn[r] + bh_n));
      float hp = hf_in[row * H + col];
      float hv = (1.0f - zg) * ng + zg * hp;
      hf_out[row * H + col] = hv;
      u16x hb = f32_to_bf16(hv);
      hb_out[row * H + col] = hb;
      y[((size_t)t * 32 + row) * H + col] = hb;
    }

    // device-wide step barrier (32 co-resident blocks)
    __syncthreads();
    if (threadIdx.x == 0) {
      __threadfence();
      atomicAdd(syncc, 1u);
      const unsigned target = (unsigned)gridDim.x * (unsigned)(t + 1);
      while (*((volatile unsigned*)syncc) < target) { }
      __threadfence();
    }
    __syncthreads();
  }
}

// --------------------------- small helper kernels ---------------------------
__global__ void cast_f32_to_bf16(const float* __restrict__ src,
                                 u16x* __restrict__ dst, long n) {
  long i = (long)blockIdx.x * blockDim.x + threadIdx.x;
  if (i < n) dst[i] = f32_to_bf16(src[i]);
}

__global__ void embed_cast(const int* __restrict__ tok, const float* __restrict__ table,
                           u16x* __restrict__ x, int Bb, int S, int E) {
  int tb = blockIdx.x;                                // t*Bb + b (time-major)
  int t = tb / Bb, b = tb % Bb;
  const float* src = table + (size_t)tok[b * S + t] * E;
  u16x* dst = x + (size_t)tb * E;
  for (int e = threadIdx.x; e < E; e += blockDim.x) dst[e] = f32_to_bf16(src[e]);
}

__global__ void init_h(const float* __restrict__ src, float* __restrict__ hf,
                       u16x* __restrict__ hb, int n) {
  int i = blockIdx.x * blockDim.x + threadIdx.x;
  if (i < n) { float v = src[i]; hf[i] = v; hb[i] = f32_to_bf16(v); }
}

// ---------------------------------------------------------------------------
extern "C" void kernel_launch(void* const* d_in, const int* in_sizes, int n_in,
                              void* d_out, int out_size, void* d_ws, size_t ws_size,
                              hipStream_t stream) {
  const int Bb = 32, S = 128, E = 512, H = 1024, V = 32000, T = 128;
  const int G3 = 3 * H;

  const int*   inputs  = (const int*)  d_in[0];
  /* d_in[1] lengths: unused by reference output */
  const float* hidden  = (const float*)d_in[2];
  const float* table   = (const float*)d_in[3];
  const float* W_ih_l0 = (const float*)d_in[4];
  const float* W_hh_l0 = (const float*)d_in[5];
  const float* b_ih_l0 = (const float*)d_in[6];
  const float* b_hh_l0 = (const float*)d_in[7];
  const float* W_ih_l1 = (const float*)d_in[8];
  const float* W_hh_l1 = (const float*)d_in[9];
  const float* b_ih_l1 = (const float*)d_in[10];
  const float* b_hh_l1 = (const float*)d_in[11];
  const float* W_out   = (const float*)d_in[12];
  const float* b_out   = (const float*)d_in[13];
  float* out = (float*)d_out;
  (void)in_sizes; (void)n_in; (void)out_size; (void)ws_size;

  char* ws = (char*)d_ws;
  size_t off = 0;
  auto take = [&](size_t bytes) -> char* {
    char* p = ws + off; off += (bytes + 255) & ~(size_t)255; return p;
  };
  u16x*  wih0 = (u16x*) take((size_t)G3 * E * 2);
  u16x*  whh0 = (u16x*) take((size_t)G3 * H * 2);
  u16x*  wih1 = (u16x*) take((size_t)G3 * H * 2);
  u16x*  whh1 = (u16x*) take((size_t)G3 * H * 2);
  u16x*  wout = (u16x*) take((size_t)V  * H * 2);
  u16x*  x0   = (u16x*) take((size_t)T * Bb * E * 2);
  float* gi   = (float*)take((size_t)T * Bb * G3 * 4);   // reused by both layers
  u16x*  y0   = (u16x*) take((size_t)T * Bb * H * 2);
  u16x*  y1   = (u16x*) take((size_t)T * Bb * H * 2);
  float* h0f  = (float*)take((size_t)Bb * H * 4);
  float* h1f  = (float*)take((size_t)Bb * H * 4);
  u16x*  h0b  = (u16x*) take((size_t)Bb * H * 2);
  u16x*  h1b  = (u16x*) take((size_t)Bb * H * 2);
  unsigned* syncc = (unsigned*)take(256);

  auto cgrid = [](long n) { return (unsigned)((n + 255) / 256); };

  // weights -> bf16 (once per call; L2-resident afterwards)
  cast_f32_to_bf16<<<cgrid((long)G3 * E), 256, 0, stream>>>(W_ih_l0, wih0, (long)G3 * E);
  cast_f32_to_bf16<<<cgrid((long)G3 * H), 256, 0, stream>>>(W_hh_l0, whh0, (long)G3 * H);
  cast_f32_to_bf16<<<cgrid((long)G3 * H), 256, 0, stream>>>(W_ih_l1, wih1, (long)G3 * H);
  cast_f32_to_bf16<<<cgrid((long)G3 * H), 256, 0, stream>>>(W_hh_l1, whh1, (long)G3 * H);
  cast_f32_to_bf16<<<cgrid((long)V  * H), 256, 0, stream>>>(W_out,   wout, (long)V  * H);

  embed_cast<<<T * Bb, 128, 0, stream>>>(inputs, table, x0, Bb, S, E);
  hipMemsetAsync(syncc, 0, 2 * sizeof(unsigned), stream);

  const size_t gemmLds = (size_t)2 * 64 * 136 * sizeof(u16x);    // 34,816 B
  const size_t gruLds  = (size_t)96 * 1040 * sizeof(u16x);       // 199,680 B < 320KB

  // ---- layer 0 ----
  init_h<<<cgrid(Bb * H), 256, 0, stream>>>(hidden, h0f, h0b, Bb * H);
  gemm_bf16<<<dim3(G3 / 128, (T * Bb) / 64), 256, gemmLds, stream>>>(
      x0, wih0, b_ih_l0, gi, T * Bb, G3, E, 0, 0);
  gru_layer_persistent<<<H / 32, 128, gruLds, stream>>>(
      whh0, b_hh_l0, gi, h0f, h1f, h0b, h1b, y0, syncc + 0, T, H);

  // ---- layer 1 ----
  init_h<<<cgrid(Bb * H), 256, 0, stream>>>(hidden + Bb * H, h0f, h0b, Bb * H);
  gemm_bf16<<<dim3(G3 / 128, (T * Bb) / 64), 256, gemmLds, stream>>>(
      y0, wih1, b_ih_l1, gi, T * Bb, G3, H, 0, 0);
  gru_layer_persistent<<<H / 32, 128, gruLds, stream>>>(
      whh1, b_hh_l1, gi, h0f, h1f, h0b, h1b, y1, syncc + 1, T, H);

  // ---- output projection, [t,b] rows scattered to [b,s], streaming stores ----
  gemm_bf16<<<dim3(V / 128, (T * Bb) / 64), 256, gemmLds, stream>>>(
      y1, wout, b_out, out, T * Bb, V, H, Bb, S);
}
